// SimpleS2VTModel_29772713296201
// MI455X (gfx1250) — matrix operand order repaired and checked by reading, not compiled, not run on hardware
//
#include <hip/hip_runtime.h>
#include <math.h>

// ---------------------------------------------------------------------------
// Types / helpers
// ---------------------------------------------------------------------------
typedef __attribute__((ext_vector_type(16))) __bf16 v16bf;
typedef __attribute__((ext_vector_type(8)))  float  v8f;
typedef __attribute__((ext_vector_type(4))) unsigned int uint32x4;
typedef __attribute__((ext_vector_type(8))) int          int32x8;
typedef __attribute__((ext_vector_type(4))) int          int32x4;

#if defined(__has_builtin)
#if __has_builtin(__builtin_amdgcn_tensor_load_to_lds) && __has_builtin(__builtin_amdgcn_s_wait_tensorcnt)
#define USE_TDM 1
#endif
#endif

#define DEV static __device__ __forceinline__

// full RNE (used in the one-time weight conversion kernel)
DEV unsigned short f2bf_rne(float f) {
  unsigned int u = __float_as_uint(f);
  u += 0x7FFFu + ((u >> 16) & 1u);
  return (unsigned short)(u >> 16);
}
// fast pack: round-half-up + one v_perm_b32  (3 VALU per 2 elements)
DEV unsigned int pack2(float lo, float hi) {
  unsigned int a = __float_as_uint(lo) + 0x8000u;
  unsigned int b = __float_as_uint(hi) + 0x8000u;
  return __builtin_amdgcn_perm(b, a, 0x07060302u);  // {b[31:16], a[31:16]}
}
union Frag { uint4 q[2]; v16bf v; };

DEV v16bf lds_frag(const unsigned short* p, int off0, int off1) {
  Frag r;
  r.q[0] = *(const uint4*)(p + off0);
  r.q[1] = *(const uint4*)(p + off1);
  return r.v;
}
DEV float sigm(float x) { return 1.0f / (1.0f + __expf(-x)); }

// ---------------------------------------------------------------------------
// One-time f32 -> bf16 (RNE) weight conversion, 2 elements / thread
// ---------------------------------------------------------------------------
__global__ void cvt_bf16_k(const float* __restrict__ src,
                           unsigned short* __restrict__ dst, int n)
{
  int i = blockIdx.x * 256 + threadIdx.x;
  int j = i * 2;
  if (j < n) {
    unsigned int lo = f2bf_rne(src[j]);
    unsigned int hi = f2bf_rne(src[j + 1]);   // all sizes here are even
    ((unsigned int*)dst)[i] = lo | (hi << 16);
  }
}

// ---------------------------------------------------------------------------
// Generic GEMM:  C[M,N] = A[M,K](f32) * W[N,K]^T(bf16) + b1[N] + b2[N]
// A converted f32->bf16 during LDS fill; W staged as straight uint4 copies.
// Software-pipelined: next chunk's global loads issue while WMMA consumes LDS.
// Workgroup tile 64x128, 8 waves, each wave 32x32 (2x2 wmma accumulators).
// Requires: M % 64 == 0, K % 32 == 0.  N arbitrary (B rows clamped; stores
// and bias guarded, so padded columns never escape).
// ---------------------------------------------------------------------------
__global__ __launch_bounds__(256)
void gemm_k(const float* __restrict__ A, const unsigned short* __restrict__ W,
            const float* __restrict__ b1, const float* __restrict__ b2,
            float* __restrict__ C, int M, int N, int K)
{
  __shared__ __align__(16) unsigned short lsA[64 * 32];
  __shared__ __align__(16) unsigned short lsB[128 * 32];

  const int tid  = threadIdx.x;
  const int lane = tid & 31;
  const int wave = tid >> 5;
  const int n0   = blockIdx.x * 128;
  const int m0   = blockIdx.y * 64;
  const int wm   = (wave >> 2) * 32;   // 0 / 32
  const int wn   = (wave & 3) * 32;    // 0 / 32 / 64 / 96

  // per-thread staging coordinates
  const int arow = tid >> 2, ak0 = (tid & 3) * 8;   // A: 8 floats
  const int brow = tid >> 1, bk0 = (tid & 1) * 16;  // B: 16 halves
  int bn = n0 + brow; if (bn > N - 1) bn = N - 1;   // clamp: branch-free pad
  const float* aSrc = A + (size_t)(m0 + arow) * K + ak0;
  const unsigned short* bSrc = W + (size_t)bn * K + bk0;

  v8f acc[2][2];
  for (int i = 0; i < 2; i++)
    for (int j = 0; j < 2; j++)
      for (int e = 0; e < 8; e++) acc[i][j][e] = 0.0f;

  // pipeline prologue: chunk 0 into registers
  float4 a0 = *(const float4*)(aSrc);
  float4 a1 = *(const float4*)(aSrc + 4);
  uint4  w0 = *(const uint4*)(bSrc);
  uint4  w1 = *(const uint4*)(bSrc + 8);

  for (int kc = 0; kc < K; kc += 32) {
    { // store staged regs -> LDS (A packs f32->bf16)
      unsigned int* d = (unsigned int*)(lsA + arow * 32 + ak0);
      d[0] = pack2(a0.x, a0.y); d[1] = pack2(a0.z, a0.w);
      d[2] = pack2(a1.x, a1.y); d[3] = pack2(a1.z, a1.w);
      uint4* db = (uint4*)(lsB + brow * 32 + bk0);
      db[0] = w0; db[1] = w1;
    }
    __syncthreads();

    if (kc + 32 < K) {  // issue next chunk's global loads; overlap with WMMA
      a0 = *(const float4*)(aSrc + kc + 32);
      a1 = *(const float4*)(aSrc + kc + 36);
      w0 = *(const uint4*)(bSrc + kc + 32);
      w1 = *(const uint4*)(bSrc + kc + 40);
      __builtin_prefetch(aSrc + kc + 64, 0, 3);
      __builtin_prefetch(bSrc + kc + 64, 0, 3);
    }

    const int kb = (lane < 16) ? 0 : 8;   // A half-group base
    const int bb = (lane < 16) ? 0 : 8;   // B half-group base
    const int ar = lane & 15;
    v16bf afr[2], bfr[2];
    afr[0] = lds_frag(lsA, (wm + ar) * 32 + kb,      (wm + ar) * 32 + kb + 16);
    afr[1] = lds_frag(lsA, (wm + 16 + ar) * 32 + kb, (wm + 16 + ar) * 32 + kb + 16);
    bfr[0] = lds_frag(lsB, (wn + ar) * 32 + bb,      (wn + ar) * 32 + bb + 16);
    bfr[1] = lds_frag(lsB, (wn + 16 + ar) * 32 + bb, (wn + 16 + ar) * 32 + bb + 16);

    for (int i = 0; i < 2; i++)
      for (int j = 0; j < 2; j++)
        acc[i][j] = __builtin_amdgcn_wmma_f32_16x16x32_bf16(
            false, afr[i], false, bfr[j], (short)0, acc[i][j], false, false);
    __syncthreads();
  }

  // store: VGPR r -> M = r + 8*(lane>=16), N = lane&15
  for (int i = 0; i < 2; i++)
    for (int j = 0; j < 2; j++) {
      int mbase = m0 + wm + i * 16 + ((lane >> 4) << 3);
      int col   = n0 + wn + j * 16 + (lane & 15);
      if (col < N) {
        float bias = (b1 ? b1[col] : 0.0f) + (b2 ? b2[col] : 0.0f);
        float* outp = C + (size_t)mbase * N + col;
        for (int r = 0; r < 8; r++) outp[(size_t)r * N] = acc[i][j][r] + bias;
      }
    }
}

// ---------------------------------------------------------------------------
// Fused LSTM step.  8 workgroups; wg owns hidden slice [hb, hb+64).
// GEMM:  g[64 x 256] = h_in[64,512] @ whh_rows^T  (rows {q*512+hb+j}, bf16)
// B tile staged by the Tensor Data Mover: 3-D tile x=32 K-cols, y=64 hidden
// rows (stride 512), z=4 gate groups (stride 512*512), fenced with
// s_wait_tensorcnt.  Then gate fusion in LDS (union overlays staging).
// ---------------------------------------------------------------------------
struct __align__(16) LstmShared {
  union {
    struct { unsigned short A[64 * 32]; unsigned short B[256 * 32]; } st;
    float G[64 * 256];
  };
};

__global__ __launch_bounds__(256)
void lstm_step_k(const float* __restrict__ xg_t, int xg_rstride,
                 const unsigned short* __restrict__ whh,   // [2048,512] bf16
                 const float* __restrict__ h_in, const float* __restrict__ c_in,
                 float* __restrict__ h_out, float* __restrict__ c_out,
                 float* __restrict__ hseq_t, int hseq_rstride)
{
  __shared__ LstmShared sh;
  const int tid  = threadIdx.x;
  const int lane = tid & 31;
  const int wave = tid >> 5;
  const int hb   = blockIdx.x * 64;   // hidden slice base
  const int wn   = wave * 32;         // local-n strip (0..224)

  const int arow = tid >> 2, ak0 = (tid & 3) * 8;
  const float* aSrc = h_in + (size_t)arow * 512 + ak0;

#if !defined(USE_TDM)
  const int q = tid >> 6, jl = tid & 63;
  const unsigned short* bSrc = whh + (size_t)(q * 512 + hb + jl) * 512;
#else
  const unsigned long long gbase =
      (unsigned long long)(size_t)(whh + (size_t)hb * 512);
  const unsigned int ldsB = (unsigned int)(size_t)&sh.st.B[0];
#endif

  v8f acc[4][2];
  for (int i = 0; i < 4; i++)
    for (int j = 0; j < 2; j++)
      for (int e = 0; e < 8; e++) acc[i][j][e] = 0.0f;

  // pipeline prologue for A
  float4 a0 = *(const float4*)(aSrc);
  float4 a1 = *(const float4*)(aSrc + 4);

  for (int kc = 0; kc < 512; kc += 32) {
#if defined(USE_TDM)
    if (wave == 0) {  // one TDM descriptor moves the whole 256x32 bf16 B tile
      unsigned long long ga = gbase + (unsigned long long)kc * 2ull;
      uint32x4 g0;
      g0[0] = 1u;                                   // count=1, user mode
      g0[1] = ldsB;                                 // lds_addr (bytes)
      g0[2] = (unsigned int)ga;                     // global_addr[31:0]
      g0[3] = (unsigned int)(ga >> 32) | (2u << 30);// addr[56:32] | type=2
      int32x8 g1;
      g1[0] = 0x10000;                              // data_size=1 (2 bytes)
      g1[1] = (int)(512u << 16);                    // tensor_dim0 = 512
      g1[2] = (int)(512u << 16);                    // tensor_dim1 = 512
      g1[3] = (int)(32u << 16);                     // tile_dim0 = 32
      g1[4] = (int)(64u | (4u << 16));              // tile_dim1=64, tile_dim2=4
      g1[5] = 512;                                  // dim0_stride[31:0]
      g1[6] = 0;                                    // dim0hi | dim1_stride lo16
      g1[7] = 4;                                    // dim1_stride (0x40000)>>16
      int32x4 g2; g2[0] = 4; g2[1] = 0; g2[2] = 0; g2[3] = 0;  // tensor_dim2=4
      int32x4 g3; g3[0] = 0; g3[1] = 0; g3[2] = 0; g3[3] = 0;
      int32x8 g4;                                   // extra group (clang-23 6-arg form)
      g4[0] = 0; g4[1] = 0; g4[2] = 0; g4[3] = 0;
      g4[4] = 0; g4[5] = 0; g4[6] = 0; g4[7] = 0;
      __builtin_amdgcn_tensor_load_to_lds(g0, g1, g2, g3, g4, 0);
    }
#else
    { // fallback: stage B = whh rows (bf16 copy, 32 halves / thread)
      const uint4* s = (const uint4*)(bSrc + kc);
      uint4* d = (uint4*)(sh.st.B + tid * 32);
      d[0] = s[0]; d[1] = s[1]; d[2] = s[2]; d[3] = s[3];
    }
#endif
    { // stage A = h_in tile 64x32 (f32 -> bf16) from pipelined regs
      unsigned int* d = (unsigned int*)(sh.st.A + arow * 32 + ak0);
      d[0] = pack2(a0.x, a0.y); d[1] = pack2(a0.z, a0.w);
      d[2] = pack2(a1.x, a1.y); d[3] = pack2(a1.z, a1.w);
    }
#if defined(USE_TDM)
    if (wave == 0) __builtin_amdgcn_s_wait_tensorcnt(0);
#endif
    __syncthreads();

    if (kc + 32 < 512) {  // next A chunk overlaps with WMMA below
      a0 = *(const float4*)(aSrc + kc + 32);
      a1 = *(const float4*)(aSrc + kc + 36);
    }

    const int kb = (lane < 16) ? 0 : 8;
    const int bb = (lane < 16) ? 0 : 8;
    const int ar = lane & 15;
    v16bf afr[4], bfr[2];
    for (int mt = 0; mt < 4; mt++) {
      int row = mt * 16 + ar;
      afr[mt] = lds_frag(sh.st.A, row * 32 + kb, row * 32 + kb + 16);
    }
    for (int nt = 0; nt < 2; nt++) {
      int row = wn + nt * 16 + ar;
      bfr[nt] = lds_frag(sh.st.B, row * 32 + bb, row * 32 + bb + 16);
    }
    for (int mt = 0; mt < 4; mt++)
      for (int nt = 0; nt < 2; nt++)
        acc[mt][nt] = __builtin_amdgcn_wmma_f32_16x16x32_bf16(
            false, afr[mt], false, bfr[nt], (short)0, acc[mt][nt], false, false);
    __syncthreads();
  }

  // dump accumulators into LDS (overlays staging; reads drained by barrier)
  for (int mt = 0; mt < 4; mt++)
    for (int nt = 0; nt < 2; nt++) {
      int mbase = mt * 16 + ((lane >> 4) << 3);
      int n     = wn + nt * 16 + (lane & 15);
      for (int r = 0; r < 8; r++) sh.G[(mbase + r) * 256 + n] = acc[mt][nt][r];
    }
  __syncthreads();

  // gate fusion: 64 batch rows x 64 hidden cols, 16 elems / thread
  for (int it = 0; it < 16; it++) {
    int fl = it * 256 + tid;
    int m  = fl >> 6;
    int hl = fl & 63;
    int j  = hb + hl;
    const float* xr = xg_t + (size_t)m * xg_rstride;
    float gi = sh.G[m * 256 +   0 + hl] + xr[j];
    float gf = sh.G[m * 256 +  64 + hl] + xr[512 + j];
    float gg = sh.G[m * 256 + 128 + hl] + xr[1024 + j];
    float go = sh.G[m * 256 + 192 + hl] + xr[1536 + j];
    float c  = sigm(gf) * c_in[(size_t)m * 512 + j] + sigm(gi) * tanhf(gg);
    float h  = sigm(go) * tanhf(c);
    c_out[(size_t)m * 512 + j] = c;
    h_out[(size_t)m * 512 + j] = h;
    hseq_t[(size_t)m * hseq_rstride + j] = h;
  }
}

// ---------------------------------------------------------------------------
// Small kernels: zero, embedding gather, attention (scores+softmax+context)
// ---------------------------------------------------------------------------
__global__ void zero_k(float* __restrict__ p, int n) {
  int i = blockIdx.x * 256 + threadIdx.x;
  if (i < n) p[i] = 0.0f;
}

__global__ void gather_emb_k(const float* __restrict__ tbl,
                             const int* __restrict__ cap,
                             float* __restrict__ outp)
{
  int i = blockIdx.x * 256 + threadIdx.x;   // over 1216*512
  if (i >= 1216 * 512) return;
  int j = i & 511;
  int r = i >> 9;            // b*19 + t
  int b = r / 19, t = r - b * 19;
  int tok = cap[b * 20 + t]; // captions[:, :-1]
  outp[i] = tbl[(size_t)tok * 512 + j];
}

__global__ __launch_bounds__(256)
void mha_attn_k(const float* __restrict__ Q, const float* __restrict__ K,
                const float* __restrict__ V, float* __restrict__ ctx)
{
  __shared__ float S[19 * 80];
  const int b = blockIdx.x >> 2;
  const int h = blockIdx.x & 3;
  const int tid = threadIdx.x;
  const float scale = 0.08838834764831845f;  // 1/sqrt(128)

  for (int p = tid; p < 19 * 80; p += 256) {
    int q = p / 80, k = p - q * 80;
    const float* qp = Q + ((size_t)(b * 19 + q)) * 512 + h * 128;
    const float* kp = K + ((size_t)(b * 80 + k)) * 512 + h * 128;
    float s = 0.0f;
    for (int d = 0; d < 128; d += 4) {
      float4 a = *(const float4*)(qp + d);
      float4 c = *(const float4*)(kp + d);
      s += a.x * c.x + a.y * c.y + a.z * c.z + a.w * c.w;
    }
    S[p] = s * scale;
  }
  __syncthreads();
  if (tid < 19) {
    float mx = -1e30f;
    for (int k = 0; k < 80; k++) mx = fmaxf(mx, S[tid * 80 + k]);
    float sum = 0.0f;
    for (int k = 0; k < 80; k++) { float e = __expf(S[tid * 80 + k] - mx); S[tid * 80 + k] = e; sum += e; }
    float inv = 1.0f / sum;
    for (int k = 0; k < 80; k++) S[tid * 80 + k] *= inv;
  }
  __syncthreads();
  for (int p = tid; p < 19 * 128; p += 256) {
    int q = p >> 7, d = p & 127;
    const float* vp = V + ((size_t)(b * 80)) * 512 + h * 128 + d;
    float o = 0.0f;
    for (int k = 0; k < 80; k++) o += S[q * 80 + k] * vp[(size_t)k * 512];
    ctx[((size_t)(b * 19 + q)) * 512 + h * 128 + d] = o;
  }
}

// ---------------------------------------------------------------------------
// Host side
// ---------------------------------------------------------------------------
static void launch_gemm(const float* A, const unsigned short* W, const float* b1,
                        const float* b2, float* C, int M, int N, int K,
                        hipStream_t s)
{
  dim3 grid((N + 127) / 128, M / 64);
  gemm_k<<<grid, 256, 0, s>>>(A, W, b1, b2, C, M, N, K);
}

extern "C" void kernel_launch(void* const* d_in, const int* in_sizes, int n_in,
                              void* d_out, int out_size, void* d_ws, size_t ws_size,
                              hipStream_t stream)
{
  (void)in_sizes; (void)n_in; (void)out_size; (void)ws_size;

  const float* vf      = (const float*)d_in[0];
  const int*   cap     = (const int*)d_in[1];
  const float* vpw     = (const float*)d_in[2];
  const float* vpb     = (const float*)d_in[3];
  const float* emb_tbl = (const float*)d_in[4];
  const float* wih[4], *whh[4], *bih[4], *bhh[4];   // e0,e1,d0,d1
  for (int l = 0; l < 4; l++) {
    wih[l] = (const float*)d_in[5 + 4 * l + 0];
    whh[l] = (const float*)d_in[5 + 4 * l + 1];
    bih[l] = (const float*)d_in[5 + 4 * l + 2];
    bhh[l] = (const float*)d_in[5 + 4 * l + 3];
  }
  const float* a_inw  = (const float*)d_in[21];
  const float* a_inb  = (const float*)d_in[22];
  const float* a_outw = (const float*)d_in[23];
  const float* a_outb = (const float*)d_in[24];
  const float* ow     = (const float*)d_in[25];
  const float* ob     = (const float*)d_in[26];
  float* logits = (float*)d_out;

  const int ME = 64 * 80;   // 5120 encoder rows
  const int MD = 64 * 19;   // 1216 decoder rows

  char*  ws  = (char*)d_ws;
  size_t off = 0;
  auto allocF = [&](size_t nElem) -> float* {
    float* p = (float*)(ws + off);
    off += ((nElem * 4) + 255) & ~(size_t)255;
    return p;
  };
  auto allocH = [&](size_t nElem) -> unsigned short* {
    unsigned short* p = (unsigned short*)(ws + off);
    off += ((nElem * 2) + 255) & ~(size_t)255;
    return p;
  };

  // activations
  float* vproj = allocF((size_t)ME * 512);    // reused as Kf later
  float* xg    = allocF((size_t)ME * 2048);
  float* enc0  = allocF((size_t)ME * 512);    // reused as Vf later
  float* enc1  = allocF((size_t)ME * 512);
  float* embb  = allocF((size_t)MD * 512);
  float* Qf    = allocF((size_t)MD * 512);
  float* ctx   = allocF((size_t)MD * 512);
  float* att   = allocF((size_t)MD * 512);
  float* dec0  = allocF((size_t)MD * 512);
  float* dec1  = allocF((size_t)MD * 512);
  float* st    = allocF((size_t)4 * 64 * 512); // h0,h1,c0,c1
  float* hbuf[2] = { st,                st + 64 * 512 };
  float* cbuf[2] = { st + 2 * 64 * 512, st + 3 * 64 * 512 };

  // bf16 weights
  unsigned short* vpw_h   = allocH((size_t)512 * 4096);
  unsigned short* wih_h[4], *whh_h[4];
  for (int l = 0; l < 4; l++) {
    wih_h[l] = allocH((size_t)2048 * 512);
    whh_h[l] = allocH((size_t)2048 * 512);
  }
  unsigned short* a_inw_h  = allocH((size_t)1536 * 512);
  unsigned short* a_outw_h = allocH((size_t)512 * 512);
  unsigned short* ow_h     = allocH((size_t)10000 * 512);

  auto cvt = [&](const float* s, unsigned short* d, size_t n) {
    cvt_bf16_k<<<((unsigned)(n / 2) + 255) / 256, 256, 0, stream>>>(s, d, (int)n);
  };
  cvt(vpw, vpw_h, (size_t)512 * 4096);
  for (int l = 0; l < 4; l++) {
    cvt(wih[l], wih_h[l], (size_t)2048 * 512);
    cvt(whh[l], whh_h[l], (size_t)2048 * 512);
  }
  cvt(a_inw, a_inw_h, (size_t)1536 * 512);
  cvt(a_outw, a_outw_h, (size_t)512 * 512);
  cvt(ow, ow_h, (size_t)10000 * 512);

  const int stN = 4 * 64 * 512;
  auto zero_states = [&]() { zero_k<<<(stN + 255) / 256, 256, 0, stream>>>(st, stN); };

  // ---- encoder ----
  launch_gemm(vf, vpw_h, vpb, nullptr, vproj, ME, 512, 4096, stream);       // vproj
  launch_gemm(vproj, wih_h[0], bih[0], bhh[0], xg, ME, 2048, 512, stream);  // xg_e0
  zero_states();
  for (int t = 0; t < 80; t++)
    lstm_step_k<<<8, 256, 0, stream>>>(xg + (size_t)t * 2048, 80 * 2048, whh_h[0],
                                       hbuf[t & 1], cbuf[t & 1],
                                       hbuf[(t + 1) & 1], cbuf[(t + 1) & 1],
                                       enc0 + (size_t)t * 512, 80 * 512);
  launch_gemm(enc0, wih_h[1], bih[1], bhh[1], xg, ME, 2048, 512, stream);   // xg_e1
  zero_states();
  for (int t = 0; t < 80; t++)
    lstm_step_k<<<8, 256, 0, stream>>>(xg + (size_t)t * 2048, 80 * 2048, whh_h[1],
                                       hbuf[t & 1], cbuf[t & 1],
                                       hbuf[(t + 1) & 1], cbuf[(t + 1) & 1],
                                       enc1 + (size_t)t * 512, 80 * 512);

  // ---- attention ----
  gather_emb_k<<<(MD * 512 + 255) / 256, 256, 0, stream>>>(emb_tbl, cap, embb);
  float* Kf = vproj;  // dead buffers reused
  float* Vf = enc0;
  launch_gemm(embb, a_inw_h,                  a_inb,        nullptr, Qf, MD, 512, 512, stream);
  launch_gemm(enc1, a_inw_h + 512 * 512,      a_inb + 512,  nullptr, Kf, ME, 512, 512, stream);
  launch_gemm(enc1, a_inw_h + 1024 * 512,     a_inb + 1024, nullptr, Vf, ME, 512, 512, stream);
  mha_attn_k<<<64 * 4, 256, 0, stream>>>(Qf, Kf, Vf, ctx);
  launch_gemm(ctx, a_outw_h, a_outb, nullptr, att, MD, 512, 512, stream);

  // ---- decoder ----
  launch_gemm(att, wih_h[2], bih[2], bhh[2], xg, MD, 2048, 512, stream);    // xg_d0
  zero_states();
  for (int t = 0; t < 19; t++)
    lstm_step_k<<<8, 256, 0, stream>>>(xg + (size_t)t * 2048, 19 * 2048, whh_h[2],
                                       hbuf[t & 1], cbuf[t & 1],
                                       hbuf[(t + 1) & 1], cbuf[(t + 1) & 1],
                                       dec0 + (size_t)t * 512, 19 * 512);
  launch_gemm(dec0, wih_h[3], bih[3], bhh[3], xg, MD, 2048, 512, stream);   // xg_d1
  zero_states();
  for (int t = 0; t < 19; t++)
    lstm_step_k<<<8, 256, 0, stream>>>(xg + (size_t)t * 2048, 19 * 2048, whh_h[3],
                                       hbuf[t & 1], cbuf[t & 1],
                                       hbuf[(t + 1) & 1], cbuf[(t + 1) & 1],
                                       dec1 + (size_t)t * 512, 19 * 512);

  // ---- logits ----
  launch_gemm(dec1, ow_h, ob, nullptr, logits, MD, 10000, 512, stream);
}
